// GroupedQueryAttention_21792664059949
// MI455X (gfx1250) — matrix operand order repaired and checked by reading, not compiled, not run on hardware
//
#include <hip/hip_runtime.h>
#include <hip/hip_bf16.h>
#include <math.h>

// ---------------------------------------------------------------------------
// Types
// ---------------------------------------------------------------------------
typedef __bf16 bf16;
typedef __attribute__((ext_vector_type(16))) __bf16 bf16x16;
typedef __attribute__((ext_vector_type(8)))  __bf16 bf16x8;
typedef __attribute__((ext_vector_type(4)))  __bf16 bf16x4;
typedef __attribute__((ext_vector_type(8)))  float  f32x8;
typedef __attribute__((ext_vector_type(4)))  float  f32x4;

typedef __attribute__((ext_vector_type(4))) unsigned int tdm_u32x4;
typedef __attribute__((ext_vector_type(8))) int          tdm_i32x8;
typedef __attribute__((ext_vector_type(4))) int          tdm_i32x4;

union Frag16 {            // 16 bf16 values per lane = one WMMA A or B operand
    bf16x16 v;
    bf16x8  h[2];
};

__device__ __forceinline__ f32x8 wmma_bf16(const Frag16& a, const Frag16& b, f32x8 c) {
    return __builtin_amdgcn_wmma_f32_16x16x32_bf16(
        /*neg_a=*/false, a.v, /*neg_b=*/false, b.v,
        /*c_mod=*/(short)0, c, /*reuse_a=*/false, /*reuse_b=*/false);
}

// ---------------------------------------------------------------------------
// Tensor Data Mover: 2D tile Global -> LDS (descriptor per ISA 08 section 8).
// flags0 packs: data_size[17:16], pad_enable[20], pad_interval[24:22],
// pad_amount[31:25]. lds_addr = low 32 bits of flat LDS pointer.
// This toolchain's builtin is the 6-arg form:
//   (u32x4 g0, i32x8 g1, i32x4 g2, i32x4 g3, i32x8 g4, i32 cpol)
// ---------------------------------------------------------------------------
__device__ __forceinline__ void tdm_load_2d(unsigned lds_addr, const void* gptr,
                                            unsigned flags0,
                                            unsigned tensor_d0, unsigned tensor_d1,
                                            unsigned tile_d0, unsigned tile_d1,
                                            unsigned stride_d0) {
    unsigned long long ga = (unsigned long long)(uintptr_t)gptr;
    tdm_u32x4 g0;
    g0[0] = 1u;                                            // count=1, user desc
    g0[1] = lds_addr;                                      // LDS byte address
    g0[2] = (unsigned)ga;                                  // global_addr[31:0]
    g0[3] = (unsigned)((ga >> 32) & 0x01FFFFFFu) | 0x80000000u; // [56:32] | type=2
    tdm_i32x8 g1;
    g1[0] = (int)flags0;                                   // mask=0 | data_size | pad cfg
    g1[1] = (int)((tensor_d0 & 0xFFFFu) << 16);            // tensor_dim0[15:0]
    g1[2] = (int)(((tensor_d0 >> 16) & 0xFFFFu) | ((tensor_d1 & 0xFFFFu) << 16));
    g1[3] = (int)(((tensor_d1 >> 16) & 0xFFFFu) | ((tile_d0 & 0xFFFFu) << 16));
    g1[4] = (int)(tile_d1 & 0xFFFFu);                      // tile_dim1, tile_dim2=0
    g1[5] = (int)stride_d0;                                // tensor_dim0_stride[31:0]
    g1[6] = 0;
    g1[7] = 0;
    tdm_i32x4 z4 = {0, 0, 0, 0};
    tdm_i32x8 z8 = {0, 0, 0, 0, 0, 0, 0, 0};
    __builtin_amdgcn_tensor_load_to_lds(g0, g1, z4, z4, z8, 0);
}

// data_size=2B (code 1) | pad_enable | pad_interval | pad_amount(4 DWORDs = 8 elems)
#define TDM_FLAGS_ROW64B  ((1u << 16) | (1u << 20) | (3u << 22) | (3u << 25)) // 16-DW rows -> stride 40 elems
#define TDM_FLAGS_ROW256B ((1u << 16) | (1u << 20) | (5u << 22) | (3u << 25)) // 64-DW rows -> stride 136 elems
#define TDM_FLAGS_ROW128B ((1u << 16) | (1u << 20) | (4u << 22) | (3u << 25)) // 32-DW rows -> stride 72 elems

__device__ __forceinline__ unsigned lds_off32(const void* p) {
    return (unsigned)(uintptr_t)p;   // flat LDS addr: low 32 bits = LDS offset
}

// ---------------------------------------------------------------------------
// Problem constants
// ---------------------------------------------------------------------------
#define D_MODEL   2048
#define KV_DIM    512
#define SEQ       2048
#define BATCH     2
#define NHEADS    32
#define NGROUPS   8
#define DEPTH     64
#define BS        (BATCH * SEQ)         // 4096 rows

// ---------------------------------------------------------------------------
// fp32 -> bf16 conversion
// ---------------------------------------------------------------------------
__global__ __launch_bounds__(256) void cvt_f32_to_bf16(const float* __restrict__ in,
                                                       bf16* __restrict__ out, int n) {
    int i = (blockIdx.x * 256 + threadIdx.x) * 4;
    if (i + 3 < n) {
        f32x4 f = *(const f32x4*)(in + i);
        bf16x4 o;
        o[0] = (bf16)f[0]; o[1] = (bf16)f[1]; o[2] = (bf16)f[2]; o[3] = (bf16)f[3];
        *(bf16x4*)(out + i) = o;
    }
}

// ---------------------------------------------------------------------------
// Tiled WMMA GEMM with TDM-staged, double-buffered LDS tiles.
//   C[M,N] = A[M,K](bf16,row) * B[K,N](bf16,row) + bias
// Block: 256 threads (8 waves), tile 128x128x32; wave tile 64x32 (4x2 frags).
// Wave 0 drives the TDM: issue tiles k+1 into buf^1, s_wait_tensorcnt(2),
// barrier, everyone computes from buf.
// ---------------------------------------------------------------------------
#define GT_M 128
#define GT_N 128
#define GT_K 32
#define A_STRIDE 40    // elems: 64B rows + 16B TDM pad
#define B_STRIDE 136   // elems: 256B rows + 16B TDM pad

template<int OUT_F32>
__global__ __launch_bounds__(256) void gemm_bf16_wmma(
    const bf16* __restrict__ A, const bf16* __restrict__ Bw,
    const float* __restrict__ bias, void* __restrict__ Cout,
    int M, int N, int K)
{
    __shared__ bf16 As[2][GT_M * A_STRIDE];   // [128][32] padded, double buffered
    __shared__ bf16 Bs[2][GT_K * B_STRIDE];   // [32][128] padded, double buffered

    const int tid  = threadIdx.x;
    const int lane = tid & 31;
    const int wave = tid >> 5;
    const int lm   = lane & 15;
    const int lh   = lane >> 4;

    const int m0 = blockIdx.y * GT_M;
    const int n0 = blockIdx.x * GT_N;
    const int wm = (wave >> 2) * 64;
    const int wn = (wave & 3) * 32;

    const unsigned asOff = lds_off32(&As[0][0]);
    const unsigned bsOff = lds_off32(&Bs[0][0]);
    const unsigned asBytes = GT_M * A_STRIDE * 2;
    const unsigned bsBytes = GT_K * B_STRIDE * 2;

    const f32x8 zero8 = {0,0,0,0,0,0,0,0};
    f32x8 acc[4][2];
    #pragma unroll
    for (int mi = 0; mi < 4; ++mi)
        #pragma unroll
        for (int ni = 0; ni < 2; ++ni) acc[mi][ni] = zero8;

    if (wave == 0) {   // prologue: DMA first k-tile into buffer 0
        tdm_load_2d(asOff, A + (size_t)m0 * K, TDM_FLAGS_ROW64B,
                    (unsigned)K, (unsigned)M, GT_K, GT_M, (unsigned)K);
        tdm_load_2d(bsOff, Bw + n0, TDM_FLAGS_ROW256B,
                    (unsigned)N, (unsigned)K, GT_N, GT_K, (unsigned)N);
    }

    int buf = 0;
    for (int k0 = 0; k0 < K; k0 += GT_K) {
        if (wave == 0) {
            if (k0 + GT_K < K) {   // prefetch next k-tile into the other buffer
                tdm_load_2d(asOff + (buf ^ 1) * asBytes,
                            A + (size_t)m0 * K + k0 + GT_K, TDM_FLAGS_ROW64B,
                            (unsigned)K, (unsigned)M, GT_K, GT_M, (unsigned)K);
                tdm_load_2d(bsOff + (buf ^ 1) * bsBytes,
                            Bw + (size_t)(k0 + GT_K) * N + n0, TDM_FLAGS_ROW256B,
                            (unsigned)N, (unsigned)K, GT_N, GT_K, (unsigned)N);
                __builtin_amdgcn_s_wait_tensorcnt(2);  // current tile pair done
            } else {
                __builtin_amdgcn_s_wait_tensorcnt(0);
            }
        }
        __syncthreads();

        const bf16* Asb = &As[buf][0];
        const bf16* Bsb = &Bs[buf][0];

        Frag16 af[4], bfr[2];
        #pragma unroll
        for (int mi = 0; mi < 4; ++mi) {        // A frag: m=lane&15, contiguous k runs
            const bf16* src = Asb + (wm + mi * 16 + lm) * A_STRIDE + lh * 8;
            af[mi].h[0] = *(const bf16x8*)(src);
            af[mi].h[1] = *(const bf16x8*)(src + 16);
        }
        #pragma unroll
        for (int ni = 0; ni < 2; ++ni) {        // B frag: n=lane&15, k=8*half+e (+8 for e>=8) (column read)
            #pragma unroll
            for (int e = 0; e < 16; ++e)
                bfr[ni].v[e] = Bsb[(lh * 8 + e + (e & 8)) * B_STRIDE + wn + ni * 16 + lm];
        }
        #pragma unroll
        for (int mi = 0; mi < 4; ++mi)
            #pragma unroll
            for (int ni = 0; ni < 2; ++ni)
                acc[mi][ni] = wmma_bf16(af[mi], bfr[ni], acc[mi][ni]);
        __syncthreads();
        buf ^= 1;
    }

    // store: C frag layout: VGPR v -> m = v + half*8, n = lane&15
    #pragma unroll
    for (int mi = 0; mi < 4; ++mi) {
        #pragma unroll
        for (int ni = 0; ni < 2; ++ni) {
            int ng = n0 + wn + ni * 16 + lm;
            float bv = bias[ng];
            #pragma unroll
            for (int v = 0; v < 8; ++v) {
                int mg = m0 + wm + mi * 16 + v + lh * 8;
                float val = acc[mi][ni][v] + bv;
                if (OUT_F32) ((float*)Cout)[(size_t)mg * N + ng] = val;
                else         ((bf16*)Cout)[(size_t)mg * N + ng] = (bf16)val;
            }
        }
    }
}

// ---------------------------------------------------------------------------
// Grouped attention. One block = one (b, head, 16-row q tile).
// Dynamic LDS: scores fp32 [16][2048] (128KB) + P bf16 [16][2048] (64KB).
// 8 waves split the 2048-key dimension (256 keys per wave).
// Phase 3 uses per-wave double-buffered TDM copies of V chunks.
// ---------------------------------------------------------------------------
#define ATTN_DYN_SMEM (131072 + 65536)
#define V_STRIDE 72                       // elems: 128B rows + 16B TDM pad
#define V_CHUNK_BYTES (32 * V_STRIDE * 2) // 4608 B per [32 key x 64 d] chunk

__global__ __launch_bounds__(256) void gqa_attn_kernel(
    const bf16* __restrict__ Qh,    // [B,S,2048]  (channel = h*64+d)
    const bf16* __restrict__ Kh,    // [B,S,512]   (channel = g*64+d)
    const bf16* __restrict__ Vh,    // [B,S,512]
    bf16* __restrict__ Ctx,         // [B,S,2048]  (channel = h*64+d)
    float* __restrict__ AttnOut)    // [B,H,S,S]
{
    extern __shared__ char smem[];
    float* sc = (float*)smem;                 // [16][2048] scores
    bf16*  P  = (bf16*)(smem + 131072);       // [16][2048] probs (bf16)
    __shared__ float rstat[16][17];
    __shared__ float rowmax[16], rowsum[16];

    const int blk = blockIdx.x;
    const int qt  = blk & 127;
    const int bh  = blk >> 7;
    const int b   = bh >> 5;
    const int h   = bh & 31;
    const int g   = h >> 2;

    const int tid  = threadIdx.x;
    const int lane = tid & 31;
    const int wave = tid >> 5;
    const int lm   = lane & 15;
    const int lh   = lane >> 4;

    const f32x8 zero8 = {0,0,0,0,0,0,0,0};

    // ---- load Q fragments for this head/tile (A operand, 16x64 = 2 frags) ----
    const size_t qbase = ((size_t)(b * SEQ + qt * 16 + lm)) * D_MODEL + h * DEPTH;
    Frag16 qa[2];
    #pragma unroll
    for (int dk = 0; dk < 2; ++dk) {
        const bf16* src = Qh + qbase + dk * 32 + lh * 8;
        qa[dk].h[0] = *(const bf16x8*)(src);
        qa[dk].h[1] = *(const bf16x8*)(src + 16);
    }

    const size_t kvbase = (size_t)b * SEQ * KV_DIM + g * DEPTH;

    // ---- phase 1: scores = scale * Q K^T, into LDS ----
    for (int nc = 0; nc < 16; ++nc) {
        int kb = wave * 256 + nc * 16;
        Frag16 kf[2];
        #pragma unroll
        for (int dk = 0; dk < 2; ++dk) {   // B frag: lane n = key, k = d
            const bf16* src = Kh + kvbase + (size_t)(kb + lm) * KV_DIM + dk * 32 + lh * 8;
            kf[dk].h[0] = *(const bf16x8*)(src);
            kf[dk].h[1] = *(const bf16x8*)(src + 16);
        }
        f32x8 c = zero8;
        c = wmma_bf16(qa[0], kf[0], c);
        c = wmma_bf16(qa[1], kf[1], c);
        #pragma unroll
        for (int v = 0; v < 8; ++v)
            sc[(v + lh * 8) * SEQ + kb + lm] = c[v] * 0.125f;   // 1/sqrt(64)
    }
    __syncthreads();

    // ---- phase 2: row softmax (16 threads per row, 128 elems each) ----
    const int row = tid >> 4;
    const int sub = tid & 15;
    float* srow = sc + row * SEQ + sub * 128;

    float mx = -3.0e38f;
    for (int j = 0; j < 128; ++j) mx = fmaxf(mx, srow[j]);
    rstat[row][sub] = mx;
    __syncthreads();
    if (sub == 0) {
        float m = rstat[row][0];
        for (int i = 1; i < 16; ++i) m = fmaxf(m, rstat[row][i]);
        rowmax[row] = m;
    }
    __syncthreads();
    float rm = rowmax[row];
    float sum = 0.f;
    for (int j = 0; j < 128; ++j) { float e = __expf(srow[j] - rm); srow[j] = e; sum += e; }
    rstat[row][sub] = sum;
    __syncthreads();
    if (sub == 0) {
        float s = 0.f;
        for (int i = 0; i < 16; ++i) s += rstat[row][i];
        rowsum[row] = s;
    }
    __syncthreads();
    {
        float inv = 1.0f / rowsum[row];
        float* arow = AttnOut + ((size_t)((b * NHEADS + h) * SEQ + qt * 16 + row)) * SEQ + sub * 128;
        bf16*  prow = P + row * SEQ + sub * 128;
        for (int j = 0; j < 128; ++j) {
            float p = srow[j] * inv;
            __builtin_nontemporal_store(p, arow + j);   // 1 GB stream: keep out of caches
            prow[j] = (bf16)p;
        }
    }
    __syncthreads();   // scores region now dead -> reused as per-wave V staging

    // ---- phase 3: ctx partial = P(16 x 256_w) @ V(256_w x 64) per wave ----
    // Each wave TDM-copies its own [32 key x 64 d] V chunks (double buffered).
    const unsigned vOff = lds_off32(smem) + wave * 2 * V_CHUNK_BYTES;
    const bf16* vchunk[2] = {
        (const bf16*)(smem + (size_t)wave * 2 * V_CHUNK_BYTES),
        (const bf16*)(smem + (size_t)wave * 2 * V_CHUNK_BYTES + V_CHUNK_BYTES)
    };

    f32x8 cf[4];
    #pragma unroll
    for (int fn = 0; fn < 4; ++fn) cf[fn] = zero8;

    tdm_load_2d(vOff, Vh + kvbase + (size_t)(wave * 256) * KV_DIM, TDM_FLAGS_ROW128B,
                KV_DIM, SEQ, DEPTH, 32, KV_DIM);

    int vb = 0;
    for (int c8 = 0; c8 < 8; ++c8) {
        if (c8 + 1 < 8) {
            tdm_load_2d(vOff + (vb ^ 1) * V_CHUNK_BYTES,
                        Vh + kvbase + (size_t)(wave * 256 + (c8 + 1) * 32) * KV_DIM,
                        TDM_FLAGS_ROW128B, KV_DIM, SEQ, DEPTH, 32, KV_DIM);
            __builtin_amdgcn_s_wait_tensorcnt(1);
        } else {
            __builtin_amdgcn_s_wait_tensorcnt(0);
        }

        int kb = wave * 256 + c8 * 32;
        Frag16 pa;         // A frag of P: m = lane&15, k = key offset in chunk
        const bf16* psrc = P + lm * SEQ + kb + lh * 8;
        pa.h[0] = *(const bf16x8*)(psrc);
        pa.h[1] = *(const bf16x8*)(psrc + 16);

        const bf16* Vs = vchunk[vb];
        #pragma unroll
        for (int fn = 0; fn < 4; ++fn) {
            Frag16 vbf;    // B frag: n = d col, k = key = 8*half + e (+8 for e>=8) (column read)
            #pragma unroll
            for (int e = 0; e < 16; ++e)
                vbf.v[e] = Vs[(lh * 8 + e + (e & 8)) * V_STRIDE + fn * 16 + lm];
            cf[fn] = wmma_bf16(pa, vbf, cf[fn]);
        }
        vb ^= 1;
    }
    __syncthreads();   // all waves done reading P before it becomes scratch

    // ---- phase 4: cross-wave reduction of ctx partials (reuse P region) ----
    float* redc = (float*)P;   // [8 waves][16 rows][64 d]
    #pragma unroll
    for (int fn = 0; fn < 4; ++fn)
        #pragma unroll
        for (int v = 0; v < 8; ++v)
            redc[(wave * 16 + (v + lh * 8)) * 64 + fn * 16 + lm] = cf[fn][v];
    __syncthreads();

    for (int i = tid; i < 16 * 64; i += 256) {
        int r = i >> 6, d = i & 63;
        float s = 0.f;
        #pragma unroll
        for (int w = 0; w < 8; ++w) s += redc[(w * 16 + r) * 64 + d];
        Ctx[((size_t)(b * SEQ + qt * 16 + r)) * D_MODEL + h * DEPTH + d] = (bf16)s;
    }
}

// ---------------------------------------------------------------------------
// Host-side launch
// ---------------------------------------------------------------------------
extern "C" void kernel_launch(void* const* d_in, const int* in_sizes, int n_in,
                              void* d_out, int out_size, void* d_ws, size_t ws_size,
                              hipStream_t stream) {
    (void)in_sizes; (void)n_in; (void)out_size; (void)ws_size;

    const float* q  = (const float*)d_in[0];
    const float* k  = (const float*)d_in[1];
    const float* v  = (const float*)d_in[2];
    const float* Wq = (const float*)d_in[3];
    const float* bq = (const float*)d_in[4];
    const float* Wk = (const float*)d_in[5];
    const float* bk = (const float*)d_in[6];
    const float* Wv = (const float*)d_in[7];
    const float* bv = (const float*)d_in[8];
    const float* Wd = (const float*)d_in[9];
    const float* bd = (const float*)d_in[10];

    float* out      = (float*)d_out;                        // [B,S,D_MODEL]
    float* attn_out = (float*)d_out + (size_t)BS * D_MODEL; // [B,H,S,S]

    // workspace layout (bf16 elements)
    const size_t SZ_X  = (size_t)BS * D_MODEL;
    const size_t SZ_WQ = (size_t)D_MODEL * D_MODEL;
    const size_t SZ_WK = (size_t)D_MODEL * KV_DIM;
    const size_t SZ_KV = (size_t)BS * KV_DIM;

    bf16* ws  = (bf16*)d_ws;
    bf16* qx  = ws;                 size_t off = SZ_X;
    bf16* kx  = ws + off;           off += SZ_X;
    bf16* vx  = ws + off;           off += SZ_X;
    bf16* Wqb = ws + off;           off += SZ_WQ;
    bf16* Wkb = ws + off;           off += SZ_WK;
    bf16* Wvb = ws + off;           off += SZ_WK;
    bf16* Wdb = ws + off;           off += SZ_WQ;
    bf16* Qh  = ws + off;           off += SZ_X;
    bf16* Kh  = ws + off;           off += SZ_KV;
    bf16* Vh  = ws + off;           off += SZ_KV;
    bf16* Ctx = ws + off;           off += SZ_X;

    // 1) fp32 -> bf16 conversions
    cvt_f32_to_bf16<<<(int)(SZ_X  / 1024), 256, 0, stream>>>(q,  qx,  (int)SZ_X);
    cvt_f32_to_bf16<<<(int)(SZ_X  / 1024), 256, 0, stream>>>(k,  kx,  (int)SZ_X);
    cvt_f32_to_bf16<<<(int)(SZ_X  / 1024), 256, 0, stream>>>(v,  vx,  (int)SZ_X);
    cvt_f32_to_bf16<<<(int)(SZ_WQ / 1024), 256, 0, stream>>>(Wq, Wqb, (int)SZ_WQ);
    cvt_f32_to_bf16<<<(int)(SZ_WK / 1024), 256, 0, stream>>>(Wk, Wkb, (int)SZ_WK);
    cvt_f32_to_bf16<<<(int)(SZ_WK / 1024), 256, 0, stream>>>(Wv, Wvb, (int)SZ_WK);
    cvt_f32_to_bf16<<<(int)(SZ_WQ / 1024), 256, 0, stream>>>(Wd, Wdb, (int)SZ_WQ);

    // 2) Q/K/V projections (bf16 WMMA + TDM staging, bf16 outputs)
    gemm_bf16_wmma<0><<<dim3(D_MODEL / GT_N, BS / GT_M), 256, 0, stream>>>(
        qx, Wqb, bq, (void*)Qh, BS, D_MODEL, D_MODEL);
    gemm_bf16_wmma<0><<<dim3(KV_DIM / GT_N, BS / GT_M), 256, 0, stream>>>(
        kx, Wkb, bk, (void*)Kh, BS, KV_DIM, D_MODEL);
    gemm_bf16_wmma<0><<<dim3(KV_DIM / GT_N, BS / GT_M), 256, 0, stream>>>(
        vx, Wvb, bv, (void*)Vh, BS, KV_DIM, D_MODEL);

    // 3) grouped attention (192 KB dynamic LDS; CDNA5 WGP has 320 KB)
    (void)hipFuncSetAttribute((const void*)gqa_attn_kernel,
                              hipFuncAttributeMaxDynamicSharedMemorySize, ATTN_DYN_SMEM);
    gqa_attn_kernel<<<BATCH * NHEADS * (SEQ / 16), 256, ATTN_DYN_SMEM, stream>>>(
        Qh, Kh, Vh, Ctx, attn_out);

    // 4) output projection (fp32 out)
    gemm_bf16_wmma<1><<<dim3(D_MODEL / GT_N, BS / GT_M), 256, 0, stream>>>(
        Ctx, Wdb, bd, (void*)out, BS, D_MODEL, D_MODEL);
}
